// Attention_65343632441539
// MI455X (gfx1250) — matrix operand order repaired and checked by reading, not compiled, not run on hardware
//
#include <hip/hip_runtime.h>
#include <hip/hip_bf16.h>

// Problem constants (from reference)
#define Bc   64
#define Sc   2048
#define Hc   512
#define NEGV (-1e9f)

// Tiling
#define ROWS_WG 128          // 8 waves * 16 rows per wave
#define XSTR    520          // bf16 elements per LDS row (512 + 8 pad -> bank spread, 16B aligned)
#define WSTR    520

typedef __attribute__((ext_vector_type(16))) __bf16        v16bf;
typedef __attribute__((ext_vector_type(8)))  float         v8f;
typedef __attribute__((ext_vector_type(4)))  unsigned int  v4u;

union Frag { v4u u[2]; v16bf v; };

static __device__ __forceinline__ unsigned short f2bf(float f) {
    unsigned int u = __float_as_uint(f);
    u += 0x7FFFu + ((u >> 16) & 1u);          // round-to-nearest-even
    return (unsigned short)(u >> 16);
}

// ---------------------------------------------------------------------------
// Kernel 1: fused  scores[b,s] = mask ? v . tanh(x Wt + b) : NEG
// Never materializes u[B,S,H]; u tiles live in WMMA accumulators only.
// ---------------------------------------------------------------------------
__global__ __launch_bounds__(256)
void scores_wmma_kernel(const float* __restrict__ x,
                        const unsigned char* __restrict__ mask,
                        const float* __restrict__ W,
                        const float* __restrict__ bias,
                        const float* __restrict__ vvec,
                        float* __restrict__ scores)
{
    __shared__ unsigned short xs[ROWS_WG * XSTR];   // 133,120 B: x slab as bf16
    __shared__ unsigned short wt[16 * WSTR];        //  16,640 B: current 16 W rows as bf16

    const int tid  = threadIdx.x;
    const int lane = tid & 31;
    const int wave = tid >> 5;
    const int nlo  = lane & 15;      // A: M row / B: N col within tile
    const int khalf = lane >> 4;     // selects K half per ISA layout
    const long rowBase = (long)blockIdx.x * ROWS_WG;

    // Stage 128 rows of x (f32 -> bf16) into LDS once; each is then reused for
    // all 32 output-channel chunks (32x data reuse out of LDS, x read once from HBM).
    for (int i = tid; i < ROWS_WG * (Hc / 4); i += 256) {
        const int r  = i / (Hc / 4);
        const int c4 = i % (Hc / 4);
        const float4 f = ((const float4*)(x + (rowBase + r) * (long)Hc))[c4];
        const int o = r * XSTR + c4 * 4;
        xs[o + 0] = f2bf(f.x); xs[o + 1] = f2bf(f.y);
        xs[o + 2] = f2bf(f.z); xs[o + 3] = f2bf(f.w);
    }

    const int mrow = wave * 16;      // this wave's 16-row tile inside the slab

    float partial[8] = {0.f, 0.f, 0.f, 0.f, 0.f, 0.f, 0.f, 0.f};

    for (int oc = 0; oc < Hc / 16; ++oc) {
        __syncthreads();
        // Stage W[oc*16 .. oc*16+15, 0..511] as bf16 (B matrix is W transposed:
        // B[k,n] = W[oc*16+n, k], handled by the per-lane fragment addressing).
        for (int i = tid; i < 16 * (Hc / 4); i += 256) {
            const int r  = i / (Hc / 4);
            const int c4 = i % (Hc / 4);
            const float4 f = ((const float4*)(W + ((size_t)(oc * 16 + r)) * Hc))[c4];
            const int o = r * WSTR + c4 * 4;
            wt[o + 0] = f2bf(f.x); wt[o + 1] = f2bf(f.y);
            wt[o + 2] = f2bf(f.z); wt[o + 3] = f2bf(f.w);
        }
        __syncthreads();

        v8f acc = {};
        for (int kc = 0; kc < Hc / 32; ++kc) {
            Frag a, b;
            // A fragment (16x32 bf16): VGPR0-3 = K 0..7 (+8 for upper lanes),
            // VGPR4-7 = K 16..23 (+8) -> two contiguous 16B LDS reads per lane.
            const int ak = kc * 32 + khalf * 8;
            a.u[0] = *(const v4u*)&xs[(mrow + nlo) * XSTR + ak];
            a.u[1] = *(const v4u*)&xs[(mrow + nlo) * XSTR + ak + 16];
            // B fragment (32x16 bf16): lane n holds K = khalf*8 + 0..7 and
            // 16 + khalf*8 + 0..7 of column n -> two 16B reads from W tile row n.
            const int bk = kc * 32 + khalf * 8;
            b.u[0] = *(const v4u*)&wt[nlo * WSTR + bk];
            b.u[1] = *(const v4u*)&wt[nlo * WSTR + bk + 16];

            acc = __builtin_amdgcn_wmma_f32_16x16x32_bf16(
                      false, a.v, false, b.v, (short)0, acc, false, false);
        }

        // Fuse: u = tanh(acc + bias[col]); partial += u * v[col]
        const int col = oc * 16 + nlo;              // C layout: N = lane & 15
        const float bc = bias[col];
        const float vc = vvec[col];
        #pragma unroll
        for (int i = 0; i < 8; ++i)
            partial[i] += tanhf(acc[i] + bc) * vc;
    }

    // C layout: VGPR i of lane L holds (M = i + 8*(L>>4), N = L&15).
    // Reduce over the 16 N-lanes of each half-wave -> scores for 16 rows.
    #pragma unroll
    for (int i = 0; i < 8; ++i) {
        float p = partial[i];
        for (int off = 8; off >= 1; off >>= 1)
            p += __shfl_xor(p, off, 16);
        partial[i] = p;
    }
    if (nlo == 0) {
        #pragma unroll
        for (int i = 0; i < 8; ++i) {
            const long row = rowBase + mrow + khalf * 8 + i;
            scores[row] = mask[row] ? partial[i] : NEGV;
        }
    }
}

// ---------------------------------------------------------------------------
// Kernel 2: per-batch softmax over S=2048; also zero-init the weighted region.
// ---------------------------------------------------------------------------
__global__ __launch_bounds__(256)
void softmax_kernel(const float* __restrict__ scores, float* __restrict__ out)
{
    __shared__ float red[256];
    const int b = blockIdx.x, t = threadIdx.x;
    const float* s = scores + (size_t)b * Sc;

    float vals[8];
    float m = -3.4e38f;
    #pragma unroll
    for (int i = 0; i < 8; ++i) { vals[i] = s[t + i * 256]; m = fmaxf(m, vals[i]); }
    red[t] = m; __syncthreads();
    for (int off = 128; off > 0; off >>= 1) {
        if (t < off) red[t] = fmaxf(red[t], red[t + off]);
        __syncthreads();
    }
    m = red[0]; __syncthreads();

    float sum = 0.f;
    #pragma unroll
    for (int i = 0; i < 8; ++i) { vals[i] = __expf(vals[i] - m); sum += vals[i]; }
    red[t] = sum; __syncthreads();
    for (int off = 128; off > 0; off >>= 1) {
        if (t < off) red[t] += red[t + off];
        __syncthreads();
    }
    const float inv = 1.f / red[0];

    float* attn = out + (size_t)Bc * Hc + (size_t)b * Sc;
    #pragma unroll
    for (int i = 0; i < 8; ++i) attn[t + i * 256] = vals[i] * inv;

    // zero the weighted_output slot for kernel 3's atomics (H = 512 = 2*256)
    out[b * Hc + t]       = 0.f;
    out[b * Hc + 256 + t] = 0.f;
}

// ---------------------------------------------------------------------------
// Kernel 3: weighted[b,h] = sum_s attn[b,s] * x[b,s,h]  (bandwidth bound:
// second 268MB stream of x; split S into 8 chunks for occupancy, f32 atomics).
// ---------------------------------------------------------------------------
__global__ __launch_bounds__(256)
void weighted_kernel(const float* __restrict__ x, float* __restrict__ out)
{
    __shared__ float aw[256];
    const int sc = blockIdx.x;          // 0..7  (chunk of 256 s values)
    const int b  = blockIdx.y;          // batch
    const int t  = threadIdx.x;

    const float* attn = out + (size_t)Bc * Hc + (size_t)b * Sc + sc * 256;
    aw[t] = attn[t];
    __syncthreads();

    const int h = t * 2;                // 256 threads * float2 = 512 = H
    float2 acc = make_float2(0.f, 0.f);
    const float* xp = x + ((size_t)b * Sc + (size_t)sc * 256) * Hc + h;
    for (int j = 0; j < 256; ++j) {
        const float  w  = aw[j];
        const float2 xv = *(const float2*)xp;
        acc.x += w * xv.x;
        acc.y += w * xv.y;
        xp += Hc;
    }
    atomicAdd(&out[b * Hc + h],     acc.x);
    atomicAdd(&out[b * Hc + h + 1], acc.y);
}

// ---------------------------------------------------------------------------
extern "C" void kernel_launch(void* const* d_in, const int* in_sizes, int n_in,
                              void* d_out, int out_size, void* d_ws, size_t ws_size,
                              hipStream_t stream)
{
    const float*         x    = (const float*)d_in[0];          // [B,S,H]
    const unsigned char* mask = (const unsigned char*)d_in[1];  // [B,S] bool (1 byte)
    const float*         W    = (const float*)d_in[2];          // [H,H]
    const float*         bia  = (const float*)d_in[3];          // [H]
    const float*         v    = (const float*)d_in[4];          // [H]
    float*               out  = (float*)d_out;                  // [B*H] ++ [B*S]
    float*               sc   = (float*)d_ws;                   // scores scratch [B*S]

    scores_wmma_kernel<<<(Bc * Sc) / ROWS_WG, 256, 0, stream>>>(x, mask, W, bia, v, sc);
    softmax_kernel<<<Bc, 256, 0, stream>>>(sc, out);
    weighted_kernel<<<dim3(8, Bc), 256, 0, stream>>>(x, out);
}
